// MultiheadSelfAttention_21612275433479
// MI455X (gfx1250) — compile-verified, matrix operand order checked by reading, not run on hardware
//
#include <hip/hip_runtime.h>
#include <hip/hip_bf16.h>

// ---------- types ----------
typedef __bf16 bf16_t;
typedef __bf16 v16bf __attribute__((ext_vector_type(16)));
typedef float  v8f   __attribute__((ext_vector_type(8)));
typedef unsigned short u16;
typedef u16 u16x8 __attribute__((ext_vector_type(8)));
typedef int gv4i __attribute__((vector_size(16)));   // generic vector: matches builtin sig

union Frag16 {
    v16bf v;
    u16x8 h[2];
};

#define D_MODEL 1024
#define NHEADS  16
#define HDIM    64
#define SEQ     2048
#define BATCH   2
#define MROWS   (BATCH * SEQ)   // 4096

#if __has_builtin(__builtin_amdgcn_global_load_async_to_lds_b128)
#define HAVE_ASYNC_LDS 1
#endif

#define AS_GLOBAL __attribute__((address_space(1)))
#define AS_LOCAL  __attribute__((address_space(3)))

__device__ __forceinline__ u16 f32_to_bf16(float f) {
    union { float f; unsigned u; } x;
    x.f = f;
    unsigned u = x.u;
    unsigned r = u + 0x7FFFu + ((u >> 16) & 1u);  // round-to-nearest-even
    return (u16)(r >> 16);
}

// Copy 16 bytes global -> LDS. Prefers gfx1250 async-to-LDS (ASYNCcnt, no VGPR
// data round trip); falls back to load+ds_store.
__device__ __forceinline__ void copy16B_g2l(const u16* __restrict__ gsrc, u16* ldst) {
#if defined(HAVE_ASYNC_LDS)
    __builtin_amdgcn_global_load_async_to_lds_b128(
        (AS_GLOBAL gv4i*)gsrc,
        (AS_LOCAL  gv4i*)ldst, 0, 0);
#else
    *(u16x8*)ldst = *(const u16x8*)gsrc;
#endif
}

__device__ __forceinline__ void wait_g2l() {
#if defined(HAVE_ASYNC_LDS)
#if __has_builtin(__builtin_amdgcn_s_wait_asynccnt)
    __builtin_amdgcn_s_wait_asynccnt(0);
#else
    asm volatile("s_wait_asynccnt 0" ::: "memory");
#endif
#endif
}

// Load a 16x32 bf16 fragment (A layout) or 32x16 (B layout, stored [n][k]) from
// a u16 array with row stride `stride` elements.
// Per ISA: lane(row=lane&15, half=lane>>4); elems 0..7 = K(half*8..+7),
// elems 8..15 = K(16+half*8..+7).
__device__ __forceinline__ Frag16 load_frag(const u16* base, int row0, int stride,
                                            int kbase, int lane) {
    int r = lane & 15;
    int h = lane >> 4;
    Frag16 f;
    const u16* p = base + (row0 + r) * stride + kbase + h * 8;
    f.h[0] = *(const u16x8*)(p);
    f.h[1] = *(const u16x8*)(p + 16);
    return f;
}

__device__ __forceinline__ v8f wmma_bf16(const Frag16& a, const Frag16& b, v8f c) {
    return __builtin_amdgcn_wmma_f32_16x16x32_bf16(
        /*neg_a=*/false, a.v, /*neg_b=*/false, b.v,
        /*c_mod=*/(short)0, c, /*reuse_a=*/false, /*reuse_b=*/false);
}

// ---------- fp32 -> bf16 conversion ----------
__global__ void cvt_bf16_kernel(const float* __restrict__ in, u16* __restrict__ out, int n) {
    for (int i = blockIdx.x * blockDim.x + threadIdx.x; i < n; i += gridDim.x * blockDim.x)
        out[i] = f32_to_bf16(in[i]);
}

// ---------- tiled GEMM: Y[m,o] = sum_d X[m,d] * W[o,d]  (x @ W^T) ----------
// Block tile 256x64, K-tile 32; 8 waves, each wave 32 rows x 64 cols
// (8 WMMA per K-step from 2 A-frags + 4 B-frags).
// MODE 0: bf16 out -> Q/K layout [b,h,s,64]   (blockIdx.y == head)
// MODE 1: bf16 out -> V^T layout [b,h,64,s]
// MODE 2: f32  out -> [m, 1024]
template <int MODE>
__global__ __launch_bounds__(256) void proj_kernel(const u16* __restrict__ X,
                                                   const u16* __restrict__ W,
                                                   void* __restrict__ Out) {
    __shared__ __align__(16) u16 As[256 * 32];  // 16 KB
    __shared__ __align__(16) u16 Bs[64 * 32];   //  4 KB

    const int tid  = threadIdx.x;
    const int lane = tid & 31;
    const int wid  = tid >> 5;           // 8 waves
    const int M0   = blockIdx.x * 256;
    const int N0   = blockIdx.y * 64;

    v8f acc[2][4] = {};

    for (int k0 = 0; k0 < D_MODEL; k0 += 32) {
        // stage A: 256 rows x 32 cols; thread t copies row t (4 x 16B)
        {
            const u16* src = X + (size_t)(M0 + tid) * D_MODEL + k0;
            u16* dst = As + tid * 32;
#pragma unroll
            for (int c = 0; c < 4; ++c)
                copy16B_g2l(src + c * 8, dst + c * 8);
            if (k0 + 32 < D_MODEL)
                __builtin_prefetch(src + 32, 0, 1);
        }
        // stage B: 64 rows(n) x 32 cols(k);  B[n][k] = W[N0+n][k0+k]
        {
            int row = tid >> 2, chunk = tid & 3;
            copy16B_g2l(W + (size_t)(N0 + row) * D_MODEL + k0 + chunk * 8,
                        Bs + row * 32 + chunk * 8);
        }
        wait_g2l();
        __syncthreads();

        Frag16 a0 = load_frag(As, wid * 32,      32, 0, lane);
        Frag16 a1 = load_frag(As, wid * 32 + 16, 32, 0, lane);
#pragma unroll
        for (int nt = 0; nt < 4; ++nt) {
            Frag16 b = load_frag(Bs, nt * 16, 32, 0, lane);
            acc[0][nt] = wmma_bf16(a0, b, acc[0][nt]);
            acc[1][nt] = wmma_bf16(a1, b, acc[1][nt]);
        }
        __syncthreads();
    }

    // epilogue: C layout: elem r <-> row r+8*half, col = lane&15
    const int nl = lane & 15;
    const int hf = lane >> 4;
#pragma unroll
    for (int rb = 0; rb < 2; ++rb) {
#pragma unroll
        for (int nt = 0; nt < 4; ++nt) {
#pragma unroll
            for (int r = 0; r < 8; ++r) {
                int m = M0 + wid * 32 + rb * 16 + r + 8 * hf;  // global row (b*S+s)
                int b = m >> 11;                               // / SEQ
                int s = m & (SEQ - 1);
                float val = acc[rb][nt][r];
                if (MODE == 0) {
                    int h = blockIdx.y;
                    int d = nt * 16 + nl;
                    ((u16*)Out)[(((size_t)b * NHEADS + h) * SEQ + s) * HDIM + d] =
                        f32_to_bf16(val);
                } else if (MODE == 1) {
                    int h = blockIdx.y;
                    int d = nt * 16 + nl;
                    ((u16*)Out)[(((size_t)b * NHEADS + h) * HDIM + d) * SEQ + s] =
                        f32_to_bf16(val);
                } else {
                    ((float*)Out)[(size_t)m * D_MODEL + N0 + nt * 16 + nl] = val;
                }
            }
        }
    }
}

// ---------- flash attention (causal) ----------
// Q,K: [b,h,s,64] bf16   Vt: [b,h,64,s] bf16   ctx out: [b*s, 1024] bf16
__global__ __launch_bounds__(256) void attn_kernel(const u16* __restrict__ Q,
                                                   const u16* __restrict__ K,
                                                   const u16* __restrict__ Vt,
                                                   u16* __restrict__ ctx) {
    __shared__ __align__(16) u16 Ks[64 * 64];       // [k_local][d]    8 KB
    __shared__ __align__(16) u16 Vs[64 * 64];       // [d][k_local]    8 KB
    __shared__ __align__(16) u16 Ps[8 * 16 * 64];   // per-wave P pad 16 KB

    const int tid  = threadIdx.x;
    const int lane = tid & 31;
    const int wid  = tid >> 5;
    const int bh   = blockIdx.y;          // b*16 + h
    const int q0   = blockIdx.x * 128;
    const int h    = bh & (NHEADS - 1);
    const int b    = bh >> 4;

    const u16* Qh = Q  + (size_t)bh * SEQ * HDIM;
    const u16* Kh = K  + (size_t)bh * SEQ * HDIM;
    const u16* Vh = Vt + (size_t)bh * HDIM * SEQ;

    const int qrow0 = q0 + wid * 16;
    const int nl = lane & 15;
    const int hf = lane >> 4;

    // Q fragments for this wave's 16 rows (K-dim = hd, two steps of 32)
    Frag16 qf[2];
    qf[0] = load_frag(Qh, qrow0, HDIM, 0,  lane);
    qf[1] = load_frag(Qh, qrow0, HDIM, 32, lane);

    v8f oacc[4] = {};
    float mrow[8], lrow[8];
#pragma unroll
    for (int r = 0; r < 8; ++r) { mrow[r] = -__builtin_huge_valf(); lrow[r] = 0.0f; }

    const int nkt = blockIdx.x * 2 + 2;   // key tiles of 64, causal bound
    const float sc = 0.125f;              // 1/sqrt(64)

    for (int kt = 0; kt < nkt; ++kt) {
        const int k0 = kt * 64;
        // stage K tile [k][d] and V tile [d][k]; thread t: row t>>2, 16B chunk t&3 (x2)
        {
            int row = tid >> 2, cb = (tid & 3) * 16;
            const u16* ks = Kh + (size_t)(k0 + row) * HDIM + cb;
            u16* kd = Ks + row * 64 + cb;
            copy16B_g2l(ks,     kd);
            copy16B_g2l(ks + 8, kd + 8);
            const u16* vs = Vh + (size_t)row * SEQ + k0 + cb;
            u16* vd = Vs + row * 64 + cb;
            copy16B_g2l(vs,     vd);
            copy16B_g2l(vs + 8, vd + 8);
        }
        wait_g2l();
        __syncthreads();

        // scores S = Q K^T  (4 key subtiles of 16)
        v8f s[4] = {};
#pragma unroll
        for (int nt = 0; nt < 4; ++nt) {
            Frag16 b0 = load_frag(Ks, nt * 16, 64, 0,  lane);
            s[nt] = wmma_bf16(qf[0], b0, s[nt]);
            Frag16 b1 = load_frag(Ks, nt * 16, 64, 32, lane);
            s[nt] = wmma_bf16(qf[1], b1, s[nt]);
        }

        // scale + causal mask + online softmax (row stats within 16-lane half)
#pragma unroll
        for (int r = 0; r < 8; ++r) {
            const int qi = q0 + wid * 16 + r + 8 * hf;
            float mx = -__builtin_huge_valf();
#pragma unroll
            for (int nt = 0; nt < 4; ++nt) {
                int ki = k0 + nt * 16 + nl;
                float v = s[nt][r] * sc;
                if (ki > qi) v = -__builtin_huge_valf();
                s[nt][r] = v;
                mx = fmaxf(mx, v);
            }
#pragma unroll
            for (int off = 1; off < 16; off <<= 1)
                mx = fmaxf(mx, __shfl_xor(mx, off, 16));
            float mnew  = fmaxf(mrow[r], mx);
            float scale = __expf(mrow[r] - mnew);
            mrow[r] = mnew;
            float rsum = 0.0f;
#pragma unroll
            for (int nt = 0; nt < 4; ++nt) {
                float p = __expf(s[nt][r] - mnew);
                s[nt][r] = p;
                rsum += p;
            }
#pragma unroll
            for (int off = 1; off < 16; off <<= 1)
                rsum += __shfl_xor(rsum, off, 16);
            lrow[r] = lrow[r] * scale + rsum;
#pragma unroll
            for (int nt = 0; nt < 4; ++nt)
                oacc[nt][r] *= scale;
        }

        // write P (bf16) to per-wave LDS pad for C->A relayout
#pragma unroll
        for (int nt = 0; nt < 4; ++nt)
#pragma unroll
            for (int r = 0; r < 8; ++r)
                Ps[(wid * 16 + r + 8 * hf) * 64 + nt * 16 + nl] = f32_to_bf16(s[nt][r]);
        __syncthreads();

        // O += P @ V   (A = P rows of this wave, B = Vs[d][k])
        Frag16 pa0 = load_frag(Ps, wid * 16, 64, 0,  lane);
        Frag16 pa1 = load_frag(Ps, wid * 16, 64, 32, lane);
#pragma unroll
        for (int nt = 0; nt < 4; ++nt) {
            Frag16 vb0 = load_frag(Vs, nt * 16, 64, 0,  lane);
            oacc[nt] = wmma_bf16(pa0, vb0, oacc[nt]);
            Frag16 vb1 = load_frag(Vs, nt * 16, 64, 32, lane);
            oacc[nt] = wmma_bf16(pa1, vb1, oacc[nt]);
        }
        __syncthreads();
    }

    // normalize and write ctx [b*s, 1024] bf16 (heads re-interleaved)
#pragma unroll
    for (int r = 0; r < 8; ++r) {
        float inv = 1.0f / lrow[r];
        int s_idx = q0 + wid * 16 + r + 8 * hf;
        size_t row = (size_t)b * SEQ + s_idx;
#pragma unroll
        for (int nt = 0; nt < 4; ++nt) {
            int col = h * HDIM + nt * 16 + nl;
            ctx[row * D_MODEL + col] = f32_to_bf16(oacc[nt][r] * inv);
        }
    }
}

// ---------- launch ----------
extern "C" void kernel_launch(void* const* d_in, const int* in_sizes, int n_in,
                              void* d_out, int out_size, void* d_ws, size_t ws_size,
                              hipStream_t stream) {
    const float* x  = (const float*)d_in[0];
    const float* Wq = (const float*)d_in[1];
    const float* Wk = (const float*)d_in[2];
    const float* Wv = (const float*)d_in[3];
    const float* Wo = (const float*)d_in[4];
    float* out = (float*)d_out;

    // workspace layout (bf16 = u16)
    const size_t NX = (size_t)MROWS * D_MODEL;       // 4096*1024
    const size_t NW = (size_t)D_MODEL * D_MODEL;     // 1024*1024
    char* ws = (char*)d_ws;
    u16* xb  = (u16*)(ws);
    u16* wqb = (u16*)(ws + NX * 2);
    u16* wkb = (u16*)(ws + NX * 2 + NW * 2);
    u16* wvb = (u16*)(ws + NX * 2 + NW * 4);
    u16* wob = (u16*)(ws + NX * 2 + NW * 6);
    u16* Qb  = (u16*)(ws + NX * 2 + NW * 8);
    u16* Kb  = (u16*)(ws + NX * 4 + NW * 8);
    u16* Vtb = (u16*)(ws + NX * 6 + NW * 8);
    u16* ctx = (u16*)(ws + NX * 8 + NW * 8);

    // 1) convert fp32 -> bf16
    cvt_bf16_kernel<<<1024, 256, 0, stream>>>(x,  xb,  (int)NX);
    cvt_bf16_kernel<<<512,  256, 0, stream>>>(Wq, wqb, (int)NW);
    cvt_bf16_kernel<<<512,  256, 0, stream>>>(Wk, wkb, (int)NW);
    cvt_bf16_kernel<<<512,  256, 0, stream>>>(Wv, wvb, (int)NW);
    cvt_bf16_kernel<<<512,  256, 0, stream>>>(Wo, wob, (int)NW);

    dim3 gridP(MROWS / 256, D_MODEL / 64);  // (16, 16)
    // 2) projections
    proj_kernel<0><<<gridP, 256, 0, stream>>>(xb, wqb, (void*)Qb);
    proj_kernel<0><<<gridP, 256, 0, stream>>>(xb, wkb, (void*)Kb);
    proj_kernel<1><<<gridP, 256, 0, stream>>>(xb, wvb, (void*)Vtb);

    // 3) causal flash attention
    dim3 gridA(SEQ / 128, BATCH * NHEADS);  // (16, 32)
    attn_kernel<<<gridA, 256, 0, stream>>>(Qb, Kb, Vtb, ctx);

    // 4) output projection (fp32 out)
    proj_kernel<2><<<gridP, 256, 0, stream>>>(ctx, wob, (void*)out);
}
